// DiTBlock_17136919511778
// MI455X (gfx1250) — matrix operand order repaired and checked by reading, not compiled, not run on hardware
//
#include <hip/hip_runtime.h>

// ---------------- problem constants ----------------
#define N_NODES 512
#define HDIM    384
#define NGR     8
#define NHEADS  6
#define DHEAD   64
#define DFFN    1536
#define MODW    2304   // 6*H

typedef __attribute__((ext_vector_type(16))) __bf16 v16bf;
typedef __attribute__((ext_vector_type(8)))  __bf16 v8bf;
typedef __attribute__((ext_vector_type(8)))  float  v8f;

__device__ __forceinline__ float siluf(float x) { return x / (1.0f + __expf(-x)); }

__device__ __forceinline__ float wred_sum(float v) {
    #pragma unroll
    for (int o = 16; o > 0; o >>= 1) v += __shfl_xor(v, o, 32);
    return v;
}
__device__ __forceinline__ float wred_max(float v) {
    #pragma unroll
    for (int o = 16; o > 0; o >>= 1) v = fmaxf(v, __shfl_xor(v, o, 32));
    return v;
}

// ---------------- elementwise converts ----------------
__global__ void k_f32_to_bf16(const float* __restrict__ s, __bf16* __restrict__ d, long long n) {
    long long i = (long long)blockIdx.x * blockDim.x + threadIdx.x;
    if (i < n) d[i] = (__bf16)s[i];
}

__global__ void k_silu_bf16(const float* __restrict__ s, __bf16* __restrict__ d, long long n) {
    long long i = (long long)blockIdx.x * blockDim.x + threadIdx.x;
    if (i < n) d[i] = (__bf16)siluf(s[i]);
}

// ---------------- graph BatchLayerNorm stats ----------------
// one block per graph; stats over columns [c0, c0+W) of src (row stride ld)
__global__ void k_bln_stats(const float* __restrict__ src, int ld, int c0, int W,
                            const int* __restrict__ batch, float eps,
                            float* __restrict__ mean, float* __restrict__ inv) {
    int g = blockIdx.x, tid = threadIdx.x;
    float s = 0.f, ss = 0.f; int nmatch = 0;
    for (int i = 0; i < N_NODES; ++i) {
        if (batch[i] != g) continue;
        ++nmatch;
        for (int f = tid; f < W; f += blockDim.x) {
            float v = src[(long long)i * ld + c0 + f];
            s += v; ss += v * v;
        }
    }
    __shared__ float sh1[256], sh2[256];
    sh1[tid] = s; sh2[tid] = ss; __syncthreads();
    for (int o = 128; o > 0; o >>= 1) {
        if (tid < o) { sh1[tid] += sh1[tid + o]; sh2[tid] += sh2[tid + o]; }
        __syncthreads();
    }
    if (tid == 0) {
        float cnt = fmaxf((float)nmatch * (float)W, 1.0f);
        float m = sh1[0] / cnt;
        float var = sh2[0] / cnt - m * m;
        mean[g] = m;
        inv[g]  = rsqrtf(var + eps);
    }
}

// out_bf16 = bln(x)*(1+mod[:,scaleCol+f]) + mod[:,shiftCol+f]
__global__ void k_modulate_bf16(const float* __restrict__ x, int ld,
                                const float* __restrict__ mean, const float* __restrict__ inv,
                                const int* __restrict__ batch, const float* __restrict__ mod,
                                int shiftCol, int scaleCol, __bf16* __restrict__ out) {
    int i = blockIdx.x, g = batch[i];
    float m = mean[g], v = inv[g];
    for (int f = threadIdx.x; f < HDIM; f += blockDim.x) {
        float val = (x[(long long)i * ld + f] - m) * v;
        val = val * (1.f + mod[(long long)i * MODW + scaleCol + f]) + mod[(long long)i * MODW + shiftCol + f];
        out[(long long)i * HDIM + f] = (__bf16)val;
    }
}

__global__ void k_modulate_f32(const float* __restrict__ x, int ld,
                               const float* __restrict__ mean, const float* __restrict__ inv,
                               const int* __restrict__ batch, const float* __restrict__ mod,
                               int shiftCol, int scaleCol, float* __restrict__ out) {
    int i = blockIdx.x, g = batch[i];
    float m = mean[g], v = inv[g];
    for (int f = threadIdx.x; f < HDIM; f += blockDim.x) {
        float val = (x[(long long)i * ld + f] - m) * v;
        val = val * (1.f + mod[(long long)i * MODW + scaleCol + f]) + mod[(long long)i * MODW + shiftCol + f];
        out[(long long)i * HDIM + f] = val;
    }
}

// f_in = bf16( bln(src)*w[f] + b[f] )
__global__ void k_bln_affine_bf16(const float* __restrict__ src,
                                  const float* __restrict__ mean, const float* __restrict__ inv,
                                  const int* __restrict__ batch,
                                  const float* __restrict__ w, const float* __restrict__ b,
                                  __bf16* __restrict__ out) {
    int i = blockIdx.x, g = batch[i];
    float m = mean[g], v = inv[g];
    for (int f = threadIdx.x; f < HDIM; f += blockDim.x) {
        float val = (src[(long long)i * HDIM + f] - m) * v * w[f] + b[f];
        out[(long long)i * HDIM + f] = (__bf16)val;
    }
}

// Q/K bln apply -> bf16; V -> bf16 transposed [H][N]
__global__ void k_qkv_norm(const float* __restrict__ qkv, const int* __restrict__ batch,
                           const float* __restrict__ mQ, const float* __restrict__ iQ,
                           const float* __restrict__ mK, const float* __restrict__ iK,
                           __bf16* __restrict__ Qbf, __bf16* __restrict__ Kbf,
                           __bf16* __restrict__ Vt) {
    int i = blockIdx.x, g = batch[i];
    float mq = mQ[g], vq = iQ[g], mk = mK[g], vk = iK[g];
    for (int f = threadIdx.x; f < HDIM; f += blockDim.x) {
        long long base = (long long)i * (3 * HDIM);
        Qbf[(long long)i * HDIM + f] = (__bf16)((qkv[base + f] - mq) * vq);
        Kbf[(long long)i * HDIM + f] = (__bf16)((qkv[base + HDIM + f] - mk) * vk);
        Vt[(long long)f * N_NODES + i] = (__bf16)qkv[base + 2 * HDIM + f];
    }
}

// residual: out = x + mod[:,gateCol+f]*y
__global__ void k_residual(const float* __restrict__ x, const float* __restrict__ mod, int gateCol,
                           const float* __restrict__ y, float* __restrict__ out) {
    int i = blockIdx.x;
    for (int f = threadIdx.x; f < HDIM; f += blockDim.x) {
        long long idx = (long long)i * HDIM + f;
        out[idx] = x[idx] + mod[(long long)i * MODW + gateCol + f] * y[idx];
    }
}

// swiglu: out_bf16[i,f] = silu(h12[i,f]) * h12[i,DFFN+f]
__global__ void k_swiglu_bf16(const float* __restrict__ h12, __bf16* __restrict__ out) {
    int i = blockIdx.x;
    for (int f = threadIdx.x; f < DFFN; f += blockDim.x) {
        float a = h12[(long long)i * (2 * DFFN) + f];
        float b = h12[(long long)i * (2 * DFFN) + DFFN + f];
        out[(long long)i * DFFN + f] = (__bf16)(siluf(a) * b);
    }
}

// ---------------- pair bias: bias[q,k] = mask + silu(Z[q,k,:])@w_pb ----------------
// one wave per (q,k); 403 MB streaming read, float4 coalesced
__global__ void __launch_bounds__(256) k_bias(const float* __restrict__ Z,
                                              const float* __restrict__ w_pb,
                                              const int* __restrict__ batch,
                                              float* __restrict__ bias) {
    int wave = threadIdx.x >> 5, lane = threadIdx.x & 31;
    long long pair = (long long)blockIdx.x * 8 + wave;     // < N*N
    int q = (int)(pair >> 9), k = (int)(pair & 511);
    const float4* zp = (const float4*)(Z + pair * HDIM);
    const float4* wp = (const float4*)w_pb;
    float acc = 0.f;
    #pragma unroll
    for (int it = 0; it < 3; ++it) {
        float4 z = zp[it * 32 + lane];
        float4 w = wp[it * 32 + lane];
        acc += siluf(z.x) * w.x + siluf(z.y) * w.y + siluf(z.z) * w.z + siluf(z.w) * w.w;
    }
    acc = wred_sum(acc);
    if (lane == 0) {
        float v;
        if (q == k)                      v = 0.0f;            // diag: same graph, bias zeroed
        else if (batch[q] == batch[k])   v = acc;
        else                             v = -__builtin_inff();
        bias[pair] = v;
    }
}

// ---------------- softmax rows -> bf16 attn ----------------
__global__ void __launch_bounds__(256) k_softmax_attn(const float* __restrict__ scores,
                                                      __bf16* __restrict__ attn) {
    int wave = threadIdx.x >> 5, lane = threadIdx.x & 31;
    int row = blockIdx.x * 8 + wave;                        // < NHEADS*N
    const float* s = scores + (long long)row * N_NODES;
    float v[16];
    float mx = -__builtin_inff();
    #pragma unroll
    for (int j = 0; j < 16; ++j) { v[j] = s[j * 32 + lane]; mx = fmaxf(mx, v[j]); }
    mx = wred_max(mx);                                      // diagonal guarantees mx finite
    float sum = 0.f;
    #pragma unroll
    for (int j = 0; j < 16; ++j) { v[j] = __expf(v[j] - mx); sum += v[j]; }
    sum = wred_sum(sum);
    float r = 1.0f / sum;
    __bf16* d = attn + (long long)row * N_NODES;
    #pragma unroll
    for (int j = 0; j < 16; ++j) d[j * 32 + lane] = (__bf16)(v[j] * r);
}

// ---------------- Z update: Zout = (Z + x⊗x) * diagmask ----------------
__global__ void __launch_bounds__(256) k_zupdate(const float* __restrict__ Z,
                                                 const float* __restrict__ x,
                                                 float* __restrict__ Zout) {
    int wave = threadIdx.x >> 5, lane = threadIdx.x & 31;
    long long pair = (long long)blockIdx.x * 8 + wave;
    int q = (int)(pair >> 9), k = (int)(pair & 511);
    const float4* zi = (const float4*)(Z + pair * HDIM);
    float4*       zo = (float4*)(Zout + pair * HDIM);
    const float4* xq = (const float4*)(x + (long long)q * HDIM);
    const float4* xk = (const float4*)(x + (long long)k * HDIM);
    bool diag = (q == k);
    #pragma unroll
    for (int it = 0; it < 3; ++it) {
        int e = it * 32 + lane;
        float4 r;
        if (diag) { r.x = r.y = r.z = r.w = 0.0f; }
        else {
            float4 z = zi[e], a = xq[e], b = xk[e];
            r.x = z.x + a.x * b.x; r.y = z.y + a.y * b.y;
            r.z = z.z + a.z * b.z; r.w = z.w + a.w * b.w;
        }
        zo[e] = r;
    }
}

// ---------------- WMMA bf16 GEMM: C = alpha * A @ B^T + addm ----------------
// A: [M x K] row-major (lda), B: [Nn x K] row-major (ldb); batched via strides.
// One wave computes one 16x16 C tile, K-loop step 32, v_wmma_f32_16x16x32_bf16.
__global__ void __launch_bounds__(256) k_gemm_nt(
        const __bf16* __restrict__ A, int lda, long long sA,
        const __bf16* __restrict__ B, int ldb, long long sB,
        float* __restrict__ C, int ldc, long long sC,
        int M, int Nn, int K, float alpha,
        const float* __restrict__ addm, int ldAdd, long long sAdd) {
    int b = blockIdx.y;
    int wave = threadIdx.x >> 5, lane = threadIdx.x & 31;
    int tilesM = M >> 4, tilesN = Nn >> 4;
    int tile = blockIdx.x * 8 + wave;
    if (tile >= tilesM * tilesN) return;                 // wave-uniform: EXEC stays full
    int tm = tile % tilesM, tn = tile / tilesM;

    const __bf16* Ab = A + b * sA;
    const __bf16* Bb = B + b * sB;

    int ra  = tm * 16 + (lane & 15);                     // A row for this lane
    int cb  = tn * 16 + (lane & 15);                     // B column (= row of B^T array)
    int kbA = (lane >> 4) << 4;                          // A: +16 for upper half-wave
    int kbB = (lane >> 4) << 4;                          // B: +16 for upper half-wave

    v8f acc = {};
    for (int kk = 0; kk < K; kk += 32) {
        const __bf16* pa = Ab + (long long)ra * lda + kk + kbA;
        v8bf a0 = *(const v8bf*)pa;                      // K = kb .. kb+7
        v8bf a1 = *(const v8bf*)(pa + 8);                // K = kb+8 .. kb+15
        v16bf av = {};
        #pragma unroll
        for (int i = 0; i < 8; ++i) { av[i] = a0[i]; av[i + 8] = a1[i]; }

        const __bf16* pb = Bb + (long long)cb * ldb + kk + kbB;
        v16bf bv = *(const v16bf*)pb;                    // 16 contiguous K

        acc = __builtin_amdgcn_wmma_f32_16x16x32_bf16(
                  false, av, false, bv, (short)0, acc, false, false);
    }

    int col   = tn * 16 + (lane & 15);
    int rbase = tm * 16 + ((lane >> 4) << 3);
    #pragma unroll
    for (int r = 0; r < 8; ++r) {
        int row = rbase + r;
        float v = alpha * acc[r];
        if (addm) v += addm[sAdd * b + (long long)row * ldAdd + col];
        C[sC * b + (long long)row * ldc + col] = v;
    }
}

// ---------------- host orchestration ----------------
static inline long long cdiv(long long a, long long b) { return (a + b - 1) / b; }

extern "C" void kernel_launch(void* const* d_in, const int* in_sizes, int n_in,
                              void* d_out, int out_size, void* d_ws, size_t ws_size,
                              hipStream_t stream) {
    (void)in_sizes; (void)n_in; (void)out_size; (void)ws_size;
    const int*   batch = (const int*)  d_in[0];
    const float* x_in  = (const float*)d_in[1];
    const float* temb  = (const float*)d_in[2];
    const float* Z     = (const float*)d_in[3];
    const float* W_ada = (const float*)d_in[4];
    const float* b_ada = (const float*)d_in[5];
    const float* W_qkv = (const float*)d_in[6];
    const float* W_out = (const float*)d_in[7];
    const float* w_pb  = (const float*)d_in[8];
    const float* W_ff1 = (const float*)d_in[9];
    const float* W_ff2 = (const float*)d_in[10];
    const float* g_ffn = (const float*)d_in[11];
    const float* b_ffn = (const float*)d_in[12];

    float* out_x = (float*)d_out;                        // [N, H]
    float* out_Z = out_x + (long long)N_NODES * HDIM;    // [N, N, H]

    // workspace carve-out (256B aligned)
    char* wsp = (char*)d_ws;
    auto alloc = [&](long long bytes) -> char* {
        char* p = wsp; wsp += (bytes + 255) & ~255LL; return p;
    };
    __bf16* Wada_bf = (__bf16*)alloc((long long)MODW * HDIM * 2);
    __bf16* Wqkv_bf = (__bf16*)alloc((long long)3 * HDIM * HDIM * 2);
    __bf16* Wout_bf = (__bf16*)alloc((long long)HDIM * HDIM * 2);
    __bf16* Wff1_bf = (__bf16*)alloc((long long)2 * DFFN * HDIM * 2);
    __bf16* Wff2_bf = (__bf16*)alloc((long long)HDIM * DFFN * 2);
    __bf16* Abf     = (__bf16*)alloc((long long)N_NODES * HDIM * 2);     // silu(t_emb)
    float*  modb    = (float*) alloc((long long)N_NODES * MODW * 4);
    __bf16* xnorm_bf= (__bf16*)alloc((long long)N_NODES * HDIM * 2);
    float*  qkv     = (float*) alloc((long long)N_NODES * 3 * HDIM * 4);
    __bf16* Qbf     = (__bf16*)alloc((long long)N_NODES * HDIM * 2);
    __bf16* Kbf     = (__bf16*)alloc((long long)N_NODES * HDIM * 2);
    __bf16* Vt      = (__bf16*)alloc((long long)HDIM * N_NODES * 2);     // [H][N]
    float*  bias    = (float*) alloc((long long)N_NODES * N_NODES * 4);
    float*  scores  = (float*) alloc((long long)NHEADS * N_NODES * N_NODES * 4);
    __bf16* attn    = (__bf16*)alloc((long long)NHEADS * N_NODES * N_NODES * 2);
    float*  outh    = (float*) alloc((long long)N_NODES * HDIM * 4);
    __bf16* outh_bf = (__bf16*)alloc((long long)N_NODES * HDIM * 2);
    float*  yb      = (float*) alloc((long long)N_NODES * HDIM * 4);
    float*  xmsa    = (float*) alloc((long long)N_NODES * HDIM * 4);
    float*  tmp     = (float*) alloc((long long)N_NODES * HDIM * 4);
    __bf16* fin_bf  = (__bf16*)alloc((long long)N_NODES * HDIM * 2);
    float*  h12     = (float*) alloc((long long)N_NODES * 2 * DFFN * 4);
    __bf16* hbf     = (__bf16*)alloc((long long)N_NODES * DFFN * 2);
    float*  ffn     = (float*) alloc((long long)N_NODES * HDIM * 4);
    float*  stats   = (float*) alloc(128 * 4);
    float *mA=stats, *iA=stats+8, *mQ=stats+16, *iQ=stats+24, *mK=stats+32, *iK=stats+40,
          *m2=stats+48, *i2=stats+56, *m3=stats+64, *i3=stats+72;

    auto conv = [&](const float* s, __bf16* d, long long n) {
        k_f32_to_bf16<<<dim3((unsigned)cdiv(n, 256)), 256, 0, stream>>>(s, d, n);
    };
    auto gemm = [&](const __bf16* A, int lda, long long sA,
                    const __bf16* B, int ldb, long long sB,
                    float* C, int ldc, long long sC,
                    int M, int Nn, int K, int nb, float alpha,
                    const float* addm, int ldAdd, long long sAdd) {
        int tiles = (M / 16) * (Nn / 16);
        dim3 g((unsigned)cdiv(tiles, 8), (unsigned)nb);
        k_gemm_nt<<<g, 256, 0, stream>>>(A, lda, sA, B, ldb, sB, C, ldc, sC,
                                         M, Nn, K, alpha, addm, ldAdd, sAdd);
    };

    // 0. weights -> bf16
    conv(W_ada, Wada_bf, (long long)MODW * HDIM);
    conv(W_qkv, Wqkv_bf, (long long)3 * HDIM * HDIM);
    conv(W_out, Wout_bf, (long long)HDIM * HDIM);
    conv(W_ff1, Wff1_bf, (long long)2 * DFFN * HDIM);
    conv(W_ff2, Wff2_bf, (long long)HDIM * DFFN);

    // 1. adaLN modulation: mod = silu(t_emb) @ W_ada^T + b_ada
    k_silu_bf16<<<dim3((unsigned)cdiv((long long)N_NODES * HDIM, 256)), 256, 0, stream>>>(
        temb, Abf, (long long)N_NODES * HDIM);
    gemm(Abf, HDIM, 0, Wada_bf, HDIM, 0, modb, MODW, 0,
         N_NODES, MODW, HDIM, 1, 1.0f, b_ada, /*ldAdd=*/0, /*sAdd=*/0);  // broadcast bias

    // 2. x_norm = bln(x)*(1+sc_msa)+sh_msa  (bf16)
    k_bln_stats<<<NGR, 256, 0, stream>>>(x_in, HDIM, 0, HDIM, batch, 1e-6f, mA, iA);
    k_modulate_bf16<<<N_NODES, 256, 0, stream>>>(x_in, HDIM, mA, iA, batch, modb, 0, HDIM, xnorm_bf);

    // 3. qkv = x_norm @ W_qkv^T
    gemm(xnorm_bf, HDIM, 0, Wqkv_bf, HDIM, 0, qkv, 3 * HDIM, 0,
         N_NODES, 3 * HDIM, HDIM, 1, 1.0f, nullptr, 0, 0);

    // 4. bln(Q), bln(K); V transposed to [H][N]
    k_bln_stats<<<NGR, 256, 0, stream>>>(qkv, 3 * HDIM, 0,        HDIM, batch, 1e-6f, mQ, iQ);
    k_bln_stats<<<NGR, 256, 0, stream>>>(qkv, 3 * HDIM, HDIM,     HDIM, batch, 1e-6f, mK, iK);
    k_qkv_norm<<<N_NODES, 256, 0, stream>>>(qkv, batch, mQ, iQ, mK, iK, Qbf, Kbf, Vt);

    // 5. pair bias from Z (403 MB streaming pass)
    k_bias<<<dim3((unsigned)(N_NODES * N_NODES / 8)), 256, 0, stream>>>(Z, w_pb, batch, bias);

    // 6. scores = Q@K^T / 8 + bias (per head; bias broadcast over heads)
    gemm(Qbf, HDIM, DHEAD, Kbf, HDIM, DHEAD, scores, N_NODES, (long long)N_NODES * N_NODES,
         N_NODES, N_NODES, DHEAD, NHEADS, 0.125f, bias, N_NODES, 0);

    // 7. softmax -> bf16 attn
    k_softmax_attn<<<dim3((unsigned)(NHEADS * N_NODES / 8)), 256, 0, stream>>>(scores, attn);

    // 8. out = attn @ V  (V^T stored [H][N]; per-head column offset in outh)
    gemm(attn, N_NODES, (long long)N_NODES * N_NODES, Vt, N_NODES, (long long)DHEAD * N_NODES,
         outh, HDIM, DHEAD, N_NODES, DHEAD, N_NODES, NHEADS, 1.0f, nullptr, 0, 0);

    // 9. y = out @ W_out^T ; x = x + g_msa*y
    conv(outh, outh_bf, (long long)N_NODES * HDIM);
    gemm(outh_bf, HDIM, 0, Wout_bf, HDIM, 0, yb, HDIM, 0,
         N_NODES, HDIM, HDIM, 1, 1.0f, nullptr, 0, 0);
    k_residual<<<N_NODES, 256, 0, stream>>>(x_in, modb, 2 * HDIM, yb, xmsa);

    // 10. f_in = bln( bln(xmsa)*(1+sc_mlp)+sh_mlp , eps=1e-5, g_ffn, b_ffn )
    k_bln_stats<<<NGR, 256, 0, stream>>>(xmsa, HDIM, 0, HDIM, batch, 1e-6f, m2, i2);
    k_modulate_f32<<<N_NODES, 256, 0, stream>>>(xmsa, HDIM, m2, i2, batch, modb, 3 * HDIM, 4 * HDIM, tmp);
    k_bln_stats<<<NGR, 256, 0, stream>>>(tmp, HDIM, 0, HDIM, batch, 1e-5f, m3, i3);
    k_bln_affine_bf16<<<N_NODES, 256, 0, stream>>>(tmp, m3, i3, batch, g_ffn, b_ffn, fin_bf);

    // 11. SwiGLU FFN
    gemm(fin_bf, HDIM, 0, Wff1_bf, HDIM, 0, h12, 2 * DFFN, 0,
         N_NODES, 2 * DFFN, HDIM, 1, 1.0f, nullptr, 0, 0);
    k_swiglu_bf16<<<N_NODES, 256, 0, stream>>>(h12, hbf);
    gemm(hbf, DFFN, 0, Wff2_bf, DFFN, 0, ffn, HDIM, 0,
         N_NODES, HDIM, DFFN, 1, 1.0f, nullptr, 0, 0);

    // 12. final x = xmsa + g_mlp*ffn  -> d_out[0 : N*H]
    k_residual<<<N_NODES, 256, 0, stream>>>(xmsa, modb, 5 * HDIM, ffn, out_x);

    // 13. Z update (403 MB read + 403 MB write) -> d_out[N*H : ]
    k_zupdate<<<dim3((unsigned)(N_NODES * N_NODES / 8)), 256, 0, stream>>>(Z, out_x, out_Z);
}
